// IMPALAComplexFruitfly_54795192762761
// MI455X (gfx1250) — matrix operand order repaired and checked by reading, not compiled, not run on hardware
//
#include <hip/hip_runtime.h>
#include <math.h>

#define B_SZ   2048
#define WIN    11
#define KDIM   4096
#define VOCABN 30522
#define NOUT   18
#define TOPK   64
#define THRESH 1e-10f

typedef _Float16 v16h __attribute__((ext_vector_type(16)));
typedef float    v8f  __attribute__((ext_vector_type(8)));

// ---------------------------------------------------------------------------
// Kernel A: per-(b,w) trig prep. trig[b*WIN+w] = {cos, sin, nz, 0}
// ---------------------------------------------------------------------------
__global__ void prep_kernel(const int* __restrict__ ids, float4* __restrict__ trig) {
  int b = blockIdx.x * blockDim.x + threadIdx.x;
  if (b >= B_SZ) return;
  int nz[WIN], posid[WIN], cum = 0;
#pragma unroll
  for (int w = 0; w < WIN; ++w) {
    int id = ids[b * WIN + w];
    nz[w] = (id != 0) ? 1 : 0;
    cum += nz[w];
    posid[w] = cum;
  }
  int sl = (cum > 0) ? cum : WIN;
  float inv = 3.14159265358979323846f / (float)sl;
#pragma unroll
  for (int w = 0; w < WIN; ++w) {
    float p = inv * (float)posid[w];
    float s, c;
    sincosf(p, &s, &c);
    trig[b * WIN + w] = make_float4(c, s, (float)nz[w], 0.0f);
  }
}

// ---------------------------------------------------------------------------
// Kernel B: heavy gather + transcendental phase (roofline-dominant).
// block = (b-tile of 256, k). Writes pf[k*B + b] coalesced. Grid ordered so
// the 8 b-tiles sharing one 244KB W-row are consecutive (L2 row reuse).
// ---------------------------------------------------------------------------
__global__ __launch_bounds__(256)
void phase_kernel(const int* __restrict__ ids,
                  const float2* __restrict__ W,      // [K][VOCAB] as float2(re,im)
                  const float4* __restrict__ trig,
                  float* __restrict__ pf) {
  int b = blockIdx.x * 256 + threadIdx.x;
  int k = blockIdx.y;
  const float2* Wrow = W + (size_t)k * VOCABN;
  float acc = 0.0f;
#pragma unroll
  for (int w = 0; w < WIN; ++w) {
    int    id = ids[b * WIN + w];
    float2 g  = Wrow[id];                       // random 8B gather within row
    float4 t  = trig[b * WIN + w];              // L2/L0 resident (352KB set)
    float wab = sqrtf(g.x * g.x + g.y * g.y);
    float pre = g.x * t.x + g.y * t.y;          // (a+bi)(c-di) real
    float pim = g.y * t.x - g.x * t.y;          // imag
    pre = (pre < THRESH && pre > -THRESH) ? THRESH : pre;
    pim = (pim < THRESH && pim > -THRESH) ? THRESH : pim;
    float ang = fabsf(atan2f(pim, pre));
    acc += (wab + ang) * t.z;
  }
  pf[(size_t)k * B_SZ + b] = acc;
}

// ---------------------------------------------------------------------------
// Kernel C: per 16-row tile: exact top-64 selection (argmax rounds, ties ->
// lower index, matching lax.top_k), then logits tile via v_wmma_f32_16x16x32_f16.
// feat is 0/1 (exact in f16); f32 accumulate.
// ---------------------------------------------------------------------------
__global__ __launch_bounds__(256)
void topk_logits_kernel(const float* __restrict__ pf,
                        const float* __restrict__ w_out,   // [NOUT][K]
                        const float* __restrict__ b_out,   // [NOUT]
                        float* __restrict__ out) {         // [B][NOUT]
  __shared__ unsigned selmask[16 * 128];   // 4096-bit mask per row
  __shared__ float red_v[8];
  __shared__ int   red_i[8];
  __shared__ int   besti_s;

  const int tid   = threadIdx.x;
  const int bbase = blockIdx.x * 16;

  for (int i = tid; i < 16 * 128; i += 256) selmask[i] = 0u;
  __syncthreads();

  // ---- selection: 16 rows, 64 argmax rounds each ----
  for (int row = 0; row < 16; ++row) {
    const int b = bbase + row;
    float vals[16];
#pragma unroll
    for (int j = 0; j < 16; ++j)
      vals[j] = pf[(size_t)(tid + 256 * j) * B_SZ + b];   // pf column, L2 hit
    unsigned taken = 0u;

    for (int it = 0; it < TOPK; ++it) {
      float bv = -3.4e38f;
      int   bi = 0x7fffffff;
#pragma unroll
      for (int j = 0; j < 16; ++j) {
        if (!((taken >> j) & 1u)) {
          int   kk = tid + 256 * j;
          float v  = vals[j];
          if (v > bv || (v == bv && kk < bi)) { bv = v; bi = kk; }
        }
      }
#pragma unroll
      for (int off = 16; off > 0; off >>= 1) {
        float ov = __shfl_xor(bv, off, 32);
        int   oi = __shfl_xor(bi, off, 32);
        if (ov > bv || (ov == bv && oi < bi)) { bv = ov; bi = oi; }
      }
      if ((tid & 31) == 0) { red_v[tid >> 5] = bv; red_i[tid >> 5] = bi; }
      __syncthreads();
      if (tid == 0) {
        float fv = red_v[0]; int fi = red_i[0];
        for (int wv = 1; wv < 8; ++wv) {
          float ov = red_v[wv]; int oi = red_i[wv];
          if (ov > fv || (ov == fv && oi < fi)) { fv = ov; fi = oi; }
        }
        besti_s = fi;
      }
      __syncthreads();
      int kwin = besti_s;
      if ((kwin & 255) == tid) {                  // owner thread marks it
        taken |= 1u << (kwin >> 8);
        selmask[row * 128 + (kwin >> 5)] |= 1u << (kwin & 31);
      }
    }
    __syncthreads();
  }

  // ---- WMMA GEMM: logits[16 x 18] = feat[16 x 4096] @ w_out^T + b_out ----
  // Wave 0 only: full-wave EXEC (WMMA requirement), tiny vs phase B.
  if (tid < 32) {
    const int lane = tid;
    const int half = lane >> 4;       // K-half selector per ISA layout
    const int n    = lane & 15;       // A: M index, B: N index
    const int n2   = n + 16;          // second N-tile (cols 16..17 valid)

    // Uniform-EXEC handling of the padded second tile: clamp row, mask value.
    const int   n2c = (n2 < NOUT) ? n2 : (NOUT - 1);
    const float msk = (n2 < NOUT) ? 1.0f : 0.0f;
    const float* wr0 = w_out + (size_t)n   * KDIM;
    const float* wr1 = w_out + (size_t)n2c * KDIM;

    v8f c0 = {0, 0, 0, 0, 0, 0, 0, 0};
    v8f c1 = {0, 0, 0, 0, 0, 0, 0, 0};

    for (int k0 = 0; k0 < KDIM; k0 += 32) {
      // A fragment: the whole 32-wide chunk lives in ONE selmask word.
      // A 16x32 layout used here (contiguous K, matches B): VGPR v -> K = 16*half + 2v (+1).
      unsigned wd = selmask[n * 128 + (k0 >> 5)];
      v16h a;
#pragma unroll
      for (int v = 0; v < 8; ++v) {
        int sh = 16 * half + 2 * v;
        a[2 * v]     = (_Float16)((wd >> sh) & 1u);
        a[2 * v + 1] = (_Float16)((wd >> (sh + 1)) & 1u);
      }
      // B fragments: lanes 0-15 cover K=k0..k0+15, lanes 16-31 K=k0+16..k0+31.
      // 16 consecutive floats per lane -> 4x float4 (global_load_b128).
      const int kb = k0 + 16 * half;                     // 64B-aligned
      const float4* p0 = (const float4*)(wr0 + kb);
      const float4* p1 = (const float4*)(wr1 + kb);
      v16h bf0, bf1;
#pragma unroll
      for (int v4 = 0; v4 < 4; ++v4) {
        float4 t0 = p0[v4];
        bf0[4 * v4 + 0] = (_Float16)t0.x;
        bf0[4 * v4 + 1] = (_Float16)t0.y;
        bf0[4 * v4 + 2] = (_Float16)t0.z;
        bf0[4 * v4 + 3] = (_Float16)t0.w;
        float4 t1 = p1[v4];
        bf1[4 * v4 + 0] = (_Float16)(t1.x * msk);
        bf1[4 * v4 + 1] = (_Float16)(t1.y * msk);
        bf1[4 * v4 + 2] = (_Float16)(t1.z * msk);
        bf1[4 * v4 + 3] = (_Float16)(t1.w * msk);
      }
      c0 = __builtin_amdgcn_wmma_f32_16x16x32_f16(false, a, false, bf0, (short)0, c0, false, false);
      c1 = __builtin_amdgcn_wmma_f32_16x16x32_f16(false, a, false, bf1, (short)0, c1, false, false);
    }

    // C/D layout: VGPR r, lanes 0-15: M=r; lanes 16-31: M=r+8; N=lane&15
#pragma unroll
    for (int r = 0; r < 8; ++r) {
      int m = r + 8 * half;
      int b = bbase + m;
      out[b * NOUT + n] = c0[r] + b_out[n];
      if (n2 < NOUT) out[b * NOUT + n2] = c1[r] + b_out[n2];
    }
  }
}

// ---------------------------------------------------------------------------
extern "C" void kernel_launch(void* const* d_in, const int* in_sizes, int n_in,
                              void* d_out, int out_size, void* d_ws, size_t ws_size,
                              hipStream_t stream) {
  const int*    ids   = (const int*)d_in[0];
  const float2* W     = (const float2*)d_in[1];    // [K][VOCAB][2] -> float2
  const float*  w_out = (const float*)d_in[2];
  const float*  b_out = (const float*)d_in[3];
  float* out = (float*)d_out;

  float*  pf   = (float*)d_ws;                                   // 32 MB
  float4* trig = (float4*)((char*)d_ws +
                           (size_t)KDIM * B_SZ * sizeof(float)); // +352 KB

  prep_kernel<<<dim3((B_SZ + 255) / 256), 256, 0, stream>>>(ids, trig);
  phase_kernel<<<dim3(B_SZ / 256, KDIM), 256, 0, stream>>>(ids, W, trig, pf);
  topk_logits_kernel<<<dim3(B_SZ / 16), 256, 0, stream>>>(pf, w_out, b_out, out);
}
